// GINnet_bin_33655363731980
// MI455X (gfx1250) — matrix operand order repaired and checked by reading, not compiled, not run on hardware
//
#include <hip/hip_runtime.h>
#include <hip/hip_bf16.h>

#define NNODES 50000
#define NEDGES 800000
#define FIN    33
#define HD     128
#define NG     64
#define BN_EPS 1e-5f

typedef __attribute__((ext_vector_type(16))) __bf16        v16bf;
typedef __attribute__((ext_vector_type(8)))  float         v8f;
typedef __attribute__((ext_vector_type(4)))  unsigned int  v4u;
typedef __attribute__((ext_vector_type(4)))  float         v4f;

union FragU { v16bf v; v4u u[2]; };

__device__ __forceinline__ unsigned short f2bf(float f) {
  // round-to-nearest-even f32 -> bf16
  unsigned int u = __float_as_uint(f);
  unsigned int r = u + 0x7FFFu + ((u >> 16) & 1u);
  return (unsigned short)(r >> 16);
}

// ---------------------------------------------------------------------------
// Convert f32 weight [K,128] -> bf16, TRANSPOSED [128 cols][128 k], zero-padded
// ---------------------------------------------------------------------------
__global__ void cvt_w_kernel(const float* __restrict__ w,
                             unsigned short* __restrict__ wt, int K) {
  int idx = blockIdx.x * blockDim.x + threadIdx.x;       // 128*128
  if (idx >= HD * HD) return;
  int c = idx >> 7;            // output column (N)
  int r = idx & 127;           // k index
  wt[idx] = (r < K) ? f2bf(w[r * HD + c]) : (unsigned short)0;
}

// ---------------------------------------------------------------------------
// Edge scatter, scalar path (layer 1, F=33): agg[dst,f] += h[src,f]
// ---------------------------------------------------------------------------
__global__ void scatter_add_kernel(const float* __restrict__ h,
                                   const int* __restrict__ src,
                                   const int* __restrict__ dst,
                                   float* __restrict__ agg,
                                   int nedges, int F) {
  int idx = blockIdx.x * blockDim.x + threadIdx.x;
  if (idx >= nedges * F) return;
  int e = idx / F;
  int f = idx - e * F;
  atomicAdd(&agg[dst[e] * F + f], h[src[e] * F + f]);
}

// ---------------------------------------------------------------------------
// Edge scatter, vectorized (F=128): each thread moves one float4 chunk.
// Consecutive lanes cover one full 512B node row -> coalesced global_load_b128.
// ---------------------------------------------------------------------------
__global__ void scatter_add4_kernel(const float* __restrict__ h,
                                    const int* __restrict__ src,
                                    const int* __restrict__ dst,
                                    float* __restrict__ agg,
                                    int nedges) {
  int idx = blockIdx.x * blockDim.x + threadIdx.x;       // nedges * 32
  if (idx >= nedges * 32) return;
  int e  = idx >> 5;
  int f4 = (idx & 31) << 2;
  const int s = src[e];
  const int d = dst[e];
  v4f val = *reinterpret_cast<const v4f*>(&h[s * HD + f4]);
  float* a = &agg[d * HD + f4];
  atomicAdd(a + 0, val[0]);
  atomicAdd(a + 1, val[1]);
  atomicAdd(a + 2, val[2]);
  atomicAdd(a + 3, val[3]);
}

// ---------------------------------------------------------------------------
// z = h + agg  -> bf16, stored row-stride 128, zero-filled for F <= k < Kpad
// ---------------------------------------------------------------------------
__global__ void add_cvt_kernel(const float* __restrict__ h,
                               const float* __restrict__ agg,
                               unsigned short* __restrict__ out,
                               int nnodes, int F, int Kpad) {
  int idx = blockIdx.x * blockDim.x + threadIdx.x;
  if (idx >= nnodes * Kpad) return;
  int n = idx / Kpad;
  int k = idx - n * Kpad;
  float z = (k < F) ? (h[n * F + k] + agg[n * F + k]) : 0.f;
  out[n * HD + k] = f2bf(z);
}

// ---------------------------------------------------------------------------
// WMMA GEMM: [32 x KPAD] bf16 tile  x  [KPAD x 128] bf16 weights -> 32x128 f32
// Block: 256 threads = 8 waves; wave w owns output columns [16w, 16w+16) and
// carries TWO accumulators (rows 0-15 and 16-31 of the tile) so the B
// fragment is loaded once and reused -> 3 ds_load_b128 per v_wmma.
// MODE 0: + bias, BatchNorm(eval), ReLU -> bf16 out
// MODE 1: + bias, ReLU                  -> f32 out
// ---------------------------------------------------------------------------
template <int MODE, int KPAD>
__global__ __launch_bounds__(256)
void gin_gemm_kernel(const unsigned short* __restrict__ A,   // [N,128] bf16
                     const unsigned short* __restrict__ Wt,  // [128,128] bf16 (W columns)
                     const float* __restrict__ bias,
                     const float* __restrict__ gamma,
                     const float* __restrict__ beta,
                     const float* __restrict__ mean,
                     const float* __restrict__ var,
                     unsigned short* __restrict__ out_bf,
                     float* __restrict__ out_f,
                     int nnodes) {
  __shared__ __align__(16) unsigned short lds_a[32 * HD];   //  8 KB
  __shared__ __align__(16) unsigned short lds_w[HD * HD];   // 32 KB

  const int tid = threadIdx.x;
  const int node_base = blockIdx.x * 32;

  __builtin_prefetch(Wt, 0, 0);   // global_prefetch_b8

  // Stage full (transposed) weight tile: 2048 x 16B
  {
    const v4u* s = reinterpret_cast<const v4u*>(Wt);
    v4u* d = reinterpret_cast<v4u*>(lds_w);
#pragma unroll
    for (int i = 0; i < 8; ++i) d[tid + i * 256] = s[tid + i * 256];
  }
  // Stage 32-row activation tile (only KPAD columns are consumed)
#pragma unroll
  for (int i = 0; i < (32 * KPAD) / 256; ++i) {
    int idx = tid + i * 256;
    int r = idx / KPAD;
    int k = idx - r * KPAD;
    int node = node_base + r;
    lds_a[r * HD + k] = (node < nnodes) ? A[node * HD + k] : (unsigned short)0;
  }
  __syncthreads();

  const int lane = tid & 31;
  const int wave = tid >> 5;
  const int hi   = lane >> 4;    // lane half (ISA §7.12.2 lane split)
  const int l16  = lane & 15;
  const int nb   = wave * 16;    // output column group

  v8f acc0 = {0.f, 0.f, 0.f, 0.f, 0.f, 0.f, 0.f, 0.f};
  v8f acc1 = acc0;
#pragma unroll
  for (int ks = 0; ks < (KPAD >> 5); ++ks) {
    const int kb = ks << 5;
    FragU af0, af1, bfr;
    // B 32x16 bf16: lane<16 holds column nb+l16, K{kb..kb+7, kb+16..kb+23}; lane>=16 +8
    const unsigned short* wp = &lds_w[(nb + l16) * HD + kb + 8 * hi];
    bfr.u[0] = *reinterpret_cast<const v4u*>(wp);
    bfr.u[1] = *reinterpret_cast<const v4u*>(wp + 16);
    // A 16x32 bf16: lane<16 holds row l16, K{kb..kb+7, kb+16..kb+23}; lane>=16 +8
    const unsigned short* ap0 = &lds_a[l16 * HD + kb + 8 * hi];
    af0.u[0] = *reinterpret_cast<const v4u*>(ap0);
    af0.u[1] = *reinterpret_cast<const v4u*>(ap0 + 16);
    const unsigned short* ap1 = &lds_a[(16 + l16) * HD + kb + 8 * hi];
    af1.u[0] = *reinterpret_cast<const v4u*>(ap1);
    af1.u[1] = *reinterpret_cast<const v4u*>(ap1 + 16);
    acc0 = __builtin_amdgcn_wmma_f32_16x16x32_bf16(false, af0.v, false, bfr.v,
                                                   (short)0, acc0, false, false);
    acc1 = __builtin_amdgcn_wmma_f32_16x16x32_bf16(false, af1.v, false, bfr.v,
                                                   (short)0, acc1, false, false);
  }

  // Epilogue: C/D layout — VGPR r holds (row r+8*hi, col nb+l16)
  const int col = nb + l16;
  float bv = bias[col];
  float sc = 1.f, sh = 0.f;
  if (MODE == 0) {
    float inv = rsqrtf(var[col] + BN_EPS);
    sc = gamma[col] * inv;
    sh = beta[col] - mean[col] * sc;
  }
#pragma unroll
  for (int half = 0; half < 2; ++half) {
    const v8f& acc = half ? acc1 : acc0;
#pragma unroll
    for (int r = 0; r < 8; ++r) {
      int node = node_base + half * 16 + r + 8 * hi;
      if (node >= nnodes) continue;
      float z = acc[r] + bv;
      if (MODE == 0) {
        z = fmaxf(z * sc + sh, 0.f);
        out_bf[node * HD + col] = f2bf(z);
      } else {
        out_f[node * HD + col] = fmaxf(z, 0.f);
      }
    }
  }
}

// ---------------------------------------------------------------------------
// Graph pooling: pooled[batch[n], {0,128,256}+f] += h{1,2,3}[n,f]
// ---------------------------------------------------------------------------
__global__ void pool_kernel(const float* __restrict__ h1,
                            const float* __restrict__ h2,
                            const float* __restrict__ h3,
                            const int* __restrict__ batch,
                            float* __restrict__ pooled, int nnodes) {
  int idx = blockIdx.x * blockDim.x + threadIdx.x;
  if (idx >= nnodes * HD) return;
  int n = idx >> 7;
  int f = idx & 127;
  int g = batch[n];
  atomicAdd(&pooled[g * (3 * HD) + f],          h1[idx]);
  atomicAdd(&pooled[g * (3 * HD) + HD + f],     h2[idx]);
  atomicAdd(&pooled[g * (3 * HD) + 2 * HD + f], h3[idx]);
}

// ---------------------------------------------------------------------------
// Head MLP: [64,384] -> relu 50 -> relu 20 -> 1 -> sigmoid  (single block)
// ---------------------------------------------------------------------------
__global__ __launch_bounds__(256)
void gin_head_kernel(const float* __restrict__ pooled,
                     const float* __restrict__ w1, const float* __restrict__ b1,
                     const float* __restrict__ w2, const float* __restrict__ b2,
                     const float* __restrict__ w3, const float* __restrict__ b3,
                     float* __restrict__ out) {
  __shared__ float t1[NG * 50];
  __shared__ float t2[NG * 20];
  const int tid = threadIdx.x;

  for (int task = tid; task < NG * 50; task += 256) {
    int g = task / 50, o = task - g * 50;
    float s = b1[o];
    const float* p = pooled + g * (3 * HD);
    for (int i = 0; i < 3 * HD; ++i) s += p[i] * w1[i * 50 + o];
    t1[task] = fmaxf(s, 0.f);
  }
  __syncthreads();
  for (int task = tid; task < NG * 20; task += 256) {
    int g = task / 20, o = task - g * 20;
    float s = b2[o];
    for (int i = 0; i < 50; ++i) s += t1[g * 50 + i] * w2[i * 20 + o];
    t2[task] = fmaxf(s, 0.f);
  }
  __syncthreads();
  for (int g = tid; g < NG; g += 256) {
    float s = b3[0];
    for (int i = 0; i < 20; ++i) s += t2[g * 20 + i] * w3[i];
    out[g] = 1.f / (1.f + __expf(-s));
  }
}

// ---------------------------------------------------------------------------
extern "C" void kernel_launch(void* const* d_in, const int* in_sizes, int n_in,
                              void* d_out, int out_size, void* d_ws, size_t ws_size,
                              hipStream_t stream) {
  (void)in_sizes; (void)n_in; (void)out_size; (void)ws_size;

  const float* x     = (const float*)d_in[0];
  const int*   ei    = (const int*)d_in[1];
  const int*   src   = ei;
  const int*   dst   = ei + NEDGES;
  const int*   batch = (const int*)d_in[2];

  const float *cw1[3], *cb1[3], *cg[3], *cbe[3], *cm[3], *cv[3], *cw2[3], *cb2[3];
  for (int li = 0; li < 3; ++li) {
    int base = 3 + li * 8;
    cw1[li] = (const float*)d_in[base + 0];
    cb1[li] = (const float*)d_in[base + 1];
    cg[li]  = (const float*)d_in[base + 2];
    cbe[li] = (const float*)d_in[base + 3];
    cm[li]  = (const float*)d_in[base + 4];
    cv[li]  = (const float*)d_in[base + 5];
    cw2[li] = (const float*)d_in[base + 6];
    cb2[li] = (const float*)d_in[base + 7];
  }
  const float* l1w = (const float*)d_in[27];
  const float* l1b = (const float*)d_in[28];
  const float* l2w = (const float*)d_in[29];
  const float* l2b = (const float*)d_in[30];
  const float* l3w = (const float*)d_in[31];
  const float* l3b = (const float*)d_in[32];

  // ---- workspace carve-up (256B aligned) ----
  char*  ws  = (char*)d_ws;
  size_t off = 0;
  auto carve = [&](size_t bytes) -> void* {
    void* p = ws + off;
    off += (bytes + 255) & ~((size_t)255);
    return p;
  };
  float*          agg    = (float*)carve((size_t)NNODES * HD * sizeof(float));
  unsigned short* zin    = (unsigned short*)carve((size_t)NNODES * HD * 2);
  unsigned short* zmid   = (unsigned short*)carve((size_t)NNODES * HD * 2);
  float*          hbuf[3];
  for (int i = 0; i < 3; ++i) hbuf[i] = (float*)carve((size_t)NNODES * HD * sizeof(float));
  unsigned short* wt1[3];
  unsigned short* wt2[3];
  for (int i = 0; i < 3; ++i) {
    wt1[i] = (unsigned short*)carve((size_t)HD * HD * 2);
    wt2[i] = (unsigned short*)carve((size_t)HD * HD * 2);
  }
  float* pooled = (float*)carve((size_t)NG * 3 * HD * sizeof(float));

  // ---- weight conversion (f32 -> bf16, transposed, zero-padded) ----
  const int wgrid = (HD * HD + 255) / 256;
  for (int li = 0; li < 3; ++li) {
    cvt_w_kernel<<<wgrid, 256, 0, stream>>>(cw1[li], wt1[li], li == 0 ? FIN : HD);
    cvt_w_kernel<<<wgrid, 256, 0, stream>>>(cw2[li], wt2[li], HD);
  }

  const int gemm_grid = (NNODES + 31) / 32;   // 1563 blocks
  const float* hin = x;
  for (int li = 0; li < 3; ++li) {
    const int Fi = (li == 0) ? FIN : HD;
    const int Kp = (li == 0) ? 64  : HD;      // pad K 33 -> 64 for layer 1

    hipMemsetAsync(agg, 0, (size_t)NNODES * Fi * sizeof(float), stream);
    if (li == 0) {
      int stot = NEDGES * FIN;
      scatter_add_kernel<<<(stot + 255) / 256, 256, 0, stream>>>(hin, src, dst,
                                                                 agg, NEDGES, FIN);
    } else {
      int stot = NEDGES * 32;
      scatter_add4_kernel<<<(stot + 255) / 256, 256, 0, stream>>>(hin, src, dst,
                                                                  agg, NEDGES);
    }
    int ctot = NNODES * Kp;
    add_cvt_kernel<<<(ctot + 255) / 256, 256, 0, stream>>>(hin, agg, zin,
                                                           NNODES, Fi, Kp);
    if (li == 0) {
      gin_gemm_kernel<0, 64><<<gemm_grid, 256, 0, stream>>>(
          zin, wt1[li], cb1[li], cg[li], cbe[li], cm[li], cv[li],
          zmid, nullptr, NNODES);
    } else {
      gin_gemm_kernel<0, 128><<<gemm_grid, 256, 0, stream>>>(
          zin, wt1[li], cb1[li], cg[li], cbe[li], cm[li], cv[li],
          zmid, nullptr, NNODES);
    }
    gin_gemm_kernel<1, 128><<<gemm_grid, 256, 0, stream>>>(
        zmid, wt2[li], cb2[li], nullptr, nullptr, nullptr, nullptr,
        nullptr, hbuf[li], NNODES);
    hin = hbuf[li];
  }

  hipMemsetAsync(pooled, 0, (size_t)NG * 3 * HD * sizeof(float), stream);
  pool_kernel<<<(NNODES * HD + 255) / 256, 256, 0, stream>>>(
      hbuf[0], hbuf[1], hbuf[2], batch, pooled, NNODES);

  gin_head_kernel<<<1, 256, 0, stream>>>(pooled, l1w, l1b, l2w, l2b, l3w, l3b,
                                         (float*)d_out);
}
